// ReduceNN_42502996361824
// MI455X (gfx1250) — compile-verified, matrix operand order checked by reading, not run on hardware
//
#include <hip/hip_runtime.h>
#include <hip/hip_bf16.h>

typedef _Float16 half_t;
typedef __attribute__((ext_vector_type(16))) _Float16 v16h;
typedef __attribute__((ext_vector_type(8)))  _Float16 v8h;
typedef __attribute__((ext_vector_type(8)))  float    v8f;

#define CDIM 64
#define BGR  128   // number of graphs

// ---------------------------------------------------------------------------
// WMMA lane mapping (CDNA5 ISA 7.12.2, 16-bit A 16x32 / B 32x16)
// lanes 0-15 : K in {0..7, 16..23};  lanes 16-31 : K in {8..15, 24..31}
// ---------------------------------------------------------------------------
__device__ __forceinline__ int wmma_koff(int lane, int j) {
    int base = (j < 8) ? j : (j + 8);
    return (lane & 16) ? (base + 8) : base;
}
__host__ __device__ __forceinline__ int wmma_koff_h(int lane, int j) {
    int base = (j < 8) ? j : (j + 8);
    return (lane & 16) ? (base + 8) : base;
}

__device__ __forceinline__ v16h cat16(v8h lo, v8h hi) {
    return __builtin_shufflevector(lo, hi, 0, 1, 2, 3, 4, 5, 6, 7,
                                   8, 9, 10, 11, 12, 13, 14, 15);
}

__device__ __forceinline__ float sigmoidf_(float x) { return 1.0f / (1.0f + __expf(-x)); }

__device__ __forceinline__ void atomicMaxF(float* addr, float val) {
    int* a = (int*)addr;
    int old = __float_as_int(*addr);
    while (__int_as_float(old) < val) {
        int assumed = old;
        old = atomicCAS(a, assumed, __float_as_int(val));
        if (old == assumed) break;
    }
}

// ---------------------------------------------------------------------------
__global__ void fill_f32(float* p, float v, int n) {
    int i = blockIdx.x * blockDim.x + threadIdx.x;
    if (i < n) p[i] = v;
}
__global__ void copy_f32(const float* s, float* d, int n) {
    int i = blockIdx.x * blockDim.x + threadIdx.x;
    if (i < n) d[i] = s[i];
}

// ---------------------------------------------------------------------------
// Pack f32 [K,N] into WMMA-fragment-major f16 layout:
//   dst[(((s*(N/16) + t)*32 + L)*16 + j] = W[s*32 + koff(L,j), t*16 + (L&15)]
// rows K..Kp zero-padded; Kp mult of 32, N mult of 16.
// ---------------------------------------------------------------------------
__global__ void pack_bfrag(const float* __restrict__ W, int K, int Nn, int Kp,
                           half_t* __restrict__ dst) {
    int total = Kp * Nn;
    int i = blockIdx.x * blockDim.x + threadIdx.x;
    if (i >= total) return;
    int j = i & 15;
    int L = (i >> 4) & 31;
    int rest = i >> 9;
    int nt16 = Nn >> 4;
    int t = rest % nt16;
    int s = rest / nt16;
    int k = s * 32 + wmma_koff(L, j);
    int n = t * 16 + (L & 15);
    dst[i] = (k < K) ? (half_t)W[(size_t)k * Nn + n] : (half_t)0.0f;
}

// Pack NNConv flattened tensor (k-major: kk = k*64 + c) into fragment layout.
//   T[kk, d] = (k<KH) ? W2[k, c*64+d] : b2[c*64+d];  N fixed = 64 (4 n-tiles)
__global__ void pack_Tfrag(const float* __restrict__ W2, const float* __restrict__ b2,
                           int KH, half_t* __restrict__ T) {
    int KT = 64 * (KH + 1);
    int total = KT * 64;
    int i = blockIdx.x * blockDim.x + threadIdx.x;
    if (i >= total) return;
    int j = i & 15;
    int L = (i >> 4) & 31;
    int rest = i >> 9;
    int t = rest & 3;
    int s = rest >> 2;
    int kk = s * 32 + wmma_koff(L, j);
    int k = kk >> 6, c = kk & 63;
    int d = t * 16 + (L & 15);
    float v = (k < KH) ? W2[(size_t)k * 4096 + c * 64 + d] : b2[c * 64 + d];
    T[i] = (half_t)v;
}

// ---------------------------------------------------------------------------
// lin0: out = relu(x[N,28] @ W[28,64] + b)
// ---------------------------------------------------------------------------
__global__ void lin0_relu(const float* __restrict__ x, const float* __restrict__ W,
                          const float* __restrict__ b, float* __restrict__ out, int N) {
    int i = blockIdx.x * blockDim.x + threadIdx.x;
    if (i >= N * 64) return;
    int r = i >> 6, c = i & 63;
    float s = b[c];
    #pragma unroll
    for (int k = 0; k < 28; ++k) s += x[r * 28 + k] * W[k * 64 + c];
    out[i] = s > 0.0f ? s : 0.0f;
}

__global__ void edge_mlp1(const float* __restrict__ ea, const float* __restrict__ W1,
                          const float* __restrict__ b1, half_t* __restrict__ h1, int E) {
    int i = blockIdx.x * blockDim.x + threadIdx.x;
    if (i >= E * 128) return;
    int e = i >> 7, j = i & 127;
    float s = b1[j];
    #pragma unroll
    for (int k = 0; k < 4; ++k) s += ea[e * 4 + k] * W1[k * 128 + j];
    h1[i] = (half_t)(s > 0.0f ? s : 0.0f);
}

__global__ void edge_mlp2(const float* __restrict__ ea, const float* __restrict__ W1,
                          const float* __restrict__ b1, half_t* __restrict__ h2, int E) {
    int i = blockIdx.x * blockDim.x + threadIdx.x;
    if (i >= E * 64) return;
    int e = i >> 6, j = i & 63;
    float s = ea[e] * W1[j] + b1[j];
    h2[i] = (half_t)(s > 0.0f ? s : 0.0f);
}

__global__ void deg_count(const int* __restrict__ dst, int E, float* __restrict__ cnt) {
    int i = blockIdx.x * blockDim.x + threadIdx.x;
    if (i < E) atomicAdd(&cnt[dst[i]], 1.0f);
}

// ---------------------------------------------------------------------------
// Generic WMMA GEMM: C[M,N] = act(A[M,Kp] @ Bfrag[Kp,N] + bias)
// A f32 row-major (lda == Kp, mult of 32, <= 256); B pre-swizzled f16 frags.
// Block = 128 thr = 4 waves; block tile 64 rows; wave tile 16 rows x N.
// ---------------------------------------------------------------------------
#define GEMM_MAXK 256
__global__ __launch_bounds__(128)
void gemm_f16(const float* __restrict__ Ag, int M, int Kp, int lda,
              const half_t* __restrict__ Bw, int N,
              const float* __restrict__ bias, int act,
              float* __restrict__ Cg, int ldc) {
    __shared__ half_t Ash[64 * GEMM_MAXK];
    int tid = threadIdx.x;
    int rowblk = blockIdx.x * 64;
    int K4 = Kp >> 2;
    for (int i = tid; i < 64 * K4; i += 128) {
        int r = i / K4, k4 = (i - r * K4) << 2;
        int gr = rowblk + r;
        float4 v = make_float4(0.f, 0.f, 0.f, 0.f);
        if (gr < M) v = *(const float4*)(Ag + (size_t)gr * lda + k4);
        half_t* q = Ash + r * Kp + k4;
        q[0] = (half_t)v.x; q[1] = (half_t)v.y; q[2] = (half_t)v.z; q[3] = (half_t)v.w;
    }
    __syncthreads();
    int wave = tid >> 5, lane = tid & 31;
    int mrow = wave * 16 + (lane & 15);
    int col  = lane & 15;
    int laneoff = (lane & 16) ? 8 : 0;
    int nt16 = N >> 4;
    const half_t* arow = Ash + mrow * Kp;
    bool fullTile = (rowblk + 64 <= M);
    for (int nb = 0; nb < N; nb += 16) {
        v8f acc = {0.f, 0.f, 0.f, 0.f, 0.f, 0.f, 0.f, 0.f};
        for (int kb = 0; kb < Kp; kb += 32) {
            v16h a = cat16(*(const v8h*)(arow + kb + laneoff),
                           *(const v8h*)(arow + kb + laneoff + 16));
            const half_t* bp = Bw + ((((size_t)(kb >> 5) * nt16) + (nb >> 4)) * 32 + lane) * 16;
            v16h b = cat16(*(const v8h*)bp, *(const v8h*)(bp + 8));
            acc = __builtin_amdgcn_wmma_f32_16x16x32_f16(false, a, false, b,
                                                         (short)0, acc, false, false);
        }
        float bv = bias ? bias[nb + col] : 0.0f;
        int rbase = rowblk + wave * 16 + ((lane & 16) ? 8 : 0);
        if (fullTile) {
            #pragma unroll
            for (int r8 = 0; r8 < 8; ++r8) {
                float v = acc[r8] + bv;
                float rv = v > 0.0f ? v : 0.0f;
                v = (act == 1) ? rv : v;
                Cg[(size_t)(rbase + r8) * ldc + nb + col] = v;
            }
        } else {
            #pragma unroll
            for (int r8 = 0; r8 < 8; ++r8) {
                if (rbase + r8 < M) {
                    float v = acc[r8] + bv;
                    float rv = v > 0.0f ? v : 0.0f;
                    v = (act == 1) ? rv : v;
                    Cg[(size_t)(rbase + r8) * ldc + nb + col] = v;
                }
            }
        }
    }
}

// ---------------------------------------------------------------------------
// Fused NNConv message kernel (k-major flattening):
//   msg[e,d] = sum_{kk} A[e,kk] * T[kk,d],  kk = k*64 + c,
//   A[e,kk] = x[src[e],c] * h'[e,k]  (h'[KH] = 1 carries the bias of the MLP)
// Within each 32-wide K-step k is constant and c is contiguous, so the A
// fragment = two 128-bit LDS loads of x + one scalar h + packed f16 muls.
// Block = 128 thr (4 waves, 64 edges); wave = 16 edges x all 64 outputs.
// ---------------------------------------------------------------------------
template <int KH>
__global__ __launch_bounds__(128)
void nnconv_msg(const float* __restrict__ X, const half_t* __restrict__ H,
                const int* __restrict__ src, const int* __restrict__ dst, int E,
                const half_t* __restrict__ T, float* __restrict__ agg) {
    constexpr int KT  = 64 * (KH + 1);  // 8256 (KH=128) / 4160 (KH=64)
    constexpr int KHS = KH + 8;         // LDS row stride (mult of 8 halves)
    __shared__ half_t xs[64 * 64];
    __shared__ half_t hh[64 * KHS];
    __shared__ int    ssh[64];
    __shared__ int    dsh[64];
    int tid = threadIdx.x;
    int ebase = blockIdx.x * 64;
    if (tid < 64) {
        int e = ebase + tid;
        ssh[tid] = (e < E) ? src[e] : -1;
        dsh[tid] = (e < E) ? dst[e] : -1;
    }
    __syncthreads();
    // stage gathered x (f32 -> f16) with float4 loads
    for (int i = tid; i < 64 * 16; i += 128) {
        int el = i >> 4, c4 = (i & 15) << 2;
        int s = ssh[el];
        float4 v = make_float4(0.f, 0.f, 0.f, 0.f);
        if (s >= 0) v = *(const float4*)(X + (size_t)s * 64 + c4);
        half_t* q = xs + el * 64 + c4;
        q[0] = (half_t)v.x; q[1] = (half_t)v.y; q[2] = (half_t)v.z; q[3] = (half_t)v.w;
    }
    // stage edge hidden (f16) with 128-bit loads; append h'[KH] = 1
    for (int i = tid; i < 64 * (KH / 8); i += 128) {
        int el = i / (KH / 8), k8 = (i - el * (KH / 8)) << 3;
        int e = ebase + el;
        v8h v = {0, 0, 0, 0, 0, 0, 0, 0};
        if (e < E) v = *(const v8h*)(H + (size_t)e * KH + k8);
        *(v8h*)(hh + el * KHS + k8) = v;
    }
    if (tid < 64) hh[tid * KHS + KH] = (ebase + tid < E) ? (half_t)1.0f : (half_t)0.0f;
    __syncthreads();

    int wave = tid >> 5, lane = tid & 31;
    int ml  = wave * 16 + (lane & 15);
    int col = lane & 15;
    int laneoff = (lane & 16) ? 8 : 0;
    const half_t* xrow = xs + ml * 64;
    const half_t* hrow = hh + ml * KHS;
    v8f acc0 = {0,0,0,0,0,0,0,0}, acc1 = {0,0,0,0,0,0,0,0};
    v8f acc2 = {0,0,0,0,0,0,0,0}, acc3 = {0,0,0,0,0,0,0,0};

    for (int kb = 0; kb < KT; kb += 32) {
        int k  = kb >> 6;             // constant over the 32-step (kb&63 + 31 < 64)
        int c0 = (kb & 63) + laneoff; // contiguous c runs for this lane
        half_t hv = hrow[k];
        v16h hs;
        #pragma unroll
        for (int j = 0; j < 16; ++j) hs[j] = hv;
        v16h xv = cat16(*(const v8h*)(xrow + c0),
                        *(const v8h*)(xrow + c0 + 16));
        v16h a = xv * hs;   // packed f16 muls
        const half_t* bp = T + ((((size_t)(kb >> 5)) * 4 * 32) + lane) * 16;
        v16h b0 = cat16(*(const v8h*)(bp +    0), *(const v8h*)(bp +    8));
        v16h b1 = cat16(*(const v8h*)(bp +  512), *(const v8h*)(bp +  520));
        v16h b2 = cat16(*(const v8h*)(bp + 1024), *(const v8h*)(bp + 1032));
        v16h b3 = cat16(*(const v8h*)(bp + 1536), *(const v8h*)(bp + 1544));
        acc0 = __builtin_amdgcn_wmma_f32_16x16x32_f16(false, a, false, b0, (short)0, acc0, false, false);
        acc1 = __builtin_amdgcn_wmma_f32_16x16x32_f16(false, a, false, b1, (short)0, acc1, false, false);
        acc2 = __builtin_amdgcn_wmma_f32_16x16x32_f16(false, a, false, b2, (short)0, acc2, false, false);
        acc3 = __builtin_amdgcn_wmma_f32_16x16x32_f16(false, a, false, b3, (short)0, acc3, false, false);
    }
    #pragma unroll
    for (int r8 = 0; r8 < 8; ++r8) {
        int el = wave * 16 + r8 + ((lane & 16) ? 8 : 0);
        int d  = dsh[el];
        if (d >= 0) {
            float* base = agg + (size_t)d * 64 + col;
            atomicAdd(base +  0, acc0[r8]);
            atomicAdd(base + 16, acc1[r8]);
            atomicAdd(base + 32, acc2[r8]);
            atomicAdd(base + 48, acc3[r8]);
        }
    }
}

// m = relu(agg/max(cnt,1) + rootterm)
__global__ void mean_root_relu(const float* __restrict__ agg, const float* __restrict__ cnt,
                               const float* __restrict__ rootterm, float* __restrict__ m, int n) {
    int i = blockIdx.x * blockDim.x + threadIdx.x;
    if (i >= n) return;
    float c = cnt[i >> 6];
    c = c > 1.0f ? c : 1.0f;
    float v = agg[i] / c + rootterm[i];
    m[i] = v > 0.0f ? v : 0.0f;
}

__global__ void gru_elem(const float* __restrict__ gx, const float* __restrict__ gh,
                         float* __restrict__ h, int n, int post) {
    int i = blockIdx.x * blockDim.x + threadIdx.x;
    if (i >= n * 64) return;
    int r = i >> 6, c = i & 63;
    const float* gxr = gx + (size_t)r * 192;
    const float* ghr = gh + (size_t)r * 192;
    float rr = sigmoidf_(gxr[c] + ghr[c]);
    float zz = sigmoidf_(gxr[64 + c] + ghr[64 + c]);
    float nn = tanhf(gxr[128 + c] + rr * ghr[128 + c]);
    float hv = (1.0f - zz) * nn + zz * h[i];
    if (post == 1) hv = hv > 0.0f ? hv : 0.0f;
    h[i] = hv;
}

__global__ void lstm_elem(const float* __restrict__ ga, const float* __restrict__ gb,
                          float* __restrict__ hl, float* __restrict__ cl, int Bn) {
    int i = blockIdx.x * blockDim.x + threadIdx.x;
    if (i >= Bn * 64) return;
    int r = i >> 6, c = i & 63;
    const float* a = ga + (size_t)r * 256;
    const float* b = gb + (size_t)r * 256;
    float gi = a[c]       + b[c];
    float gf = a[64 + c]  + b[64 + c];
    float gg = a[128 + c] + b[128 + c];
    float go = a[192 + c] + b[192 + c];
    float cn = sigmoidf_(gf) * cl[i] + sigmoidf_(gi) * tanhf(gg);
    cl[i] = cn;
    hl[i] = sigmoidf_(go) * tanhf(cn);
}

__global__ void scatter_pool(const float* __restrict__ out, const int* __restrict__ sid,
                             const int* __restrict__ tid, int Np, float* __restrict__ rg0) {
    int i = blockIdx.x * blockDim.x + threadIdx.x;
    if (i >= Np * 64) return;
    int p = i >> 6, c = i & 63;
    atomicAdd(&rg0[(size_t)tid[p] * 64 + c], out[(size_t)sid[p] * 64 + c]);
}

__global__ void dot64(const float* __restrict__ V, const float* __restrict__ w,
                      float* __restrict__ out, int n) {
    int i = blockIdx.x * blockDim.x + threadIdx.x;
    if (i >= n) return;
    float s = 0.0f;
    #pragma unroll
    for (int c = 0; c < 64; ++c) s += V[(size_t)i * 64 + c] * w[c];
    out[i] = s;
}

__global__ void gat_logit(const float* __restrict__ as_, const float* __restrict__ ad_,
                          const int* __restrict__ sid, const int* __restrict__ tid,
                          float* __restrict__ logit, int n) {
    int i = blockIdx.x * blockDim.x + threadIdx.x;
    if (i >= n) return;
    float l = as_[sid[i]] + ad_[tid[i]];
    logit[i] = l > 0.0f ? l : 0.01f * l;
}

__global__ void seg_max_k(const float* __restrict__ v, const int* __restrict__ idx,
                          float* __restrict__ mx, int n) {
    int i = blockIdx.x * blockDim.x + threadIdx.x;
    if (i < n) atomicMaxF(&mx[idx[i]], v[i]);
}

__global__ void seg_exp_sum(const float* __restrict__ v, const int* __restrict__ idx,
                            const float* __restrict__ mx, float* __restrict__ den,
                            float* __restrict__ ev, int n) {
    int i = blockIdx.x * blockDim.x + threadIdx.x;
    if (i >= n) return;
    float e = __expf(v[i] - mx[idx[i]]);
    ev[i] = e;
    atomicAdd(&den[idx[i]], e);
}

__global__ void gat_agg(const float* __restrict__ ev, const float* __restrict__ den,
                        const int* __restrict__ tid, const int* __restrict__ sid,
                        const float* __restrict__ ws, float* __restrict__ g, int Np) {
    int i = blockIdx.x * blockDim.x + threadIdx.x;
    if (i >= Np * 64) return;
    int p = i >> 6, c = i & 63;
    int t = tid[p];
    float d = den[t];
    d = (d == 0.0f) ? 1.0f : d;
    float coef = ev[p] / d;
    atomicAdd(&g[(size_t)t * 64 + c], coef * ws[(size_t)sid[p] * 64 + c]);
}

__global__ void gat_bias_elu(float* __restrict__ g, const float* __restrict__ bias, int n) {
    int i = blockIdx.x * blockDim.x + threadIdx.x;
    if (i >= n) return;
    float v = g[i] + bias[i & 63];
    g[i] = v > 0.0f ? v : (__expf(v) - 1.0f);
}

__global__ void build_Apad(const float* __restrict__ rg, const float* __restrict__ rgx,
                           float* __restrict__ Ap, int R) {
    int i = blockIdx.x * blockDim.x + threadIdx.x;
    if (i >= R * 96) return;
    int r = i / 96, k = i - r * 96;
    float v = 0.0f;
    if (k < 64) v = rg[(size_t)r * 64 + k];
    else if (k < 82) v = rgx[(size_t)r * 18 + (k - 64)];
    Ap[i] = v;
}

__global__ void s2s_edot(const float* __restrict__ rg, const float* __restrict__ hl,
                         const int* __restrict__ batch, float* __restrict__ e, int R) {
    int i = blockIdx.x * blockDim.x + threadIdx.x;
    if (i >= R) return;
    const float* a = rg + (size_t)i * 64;
    const float* b = hl + (size_t)batch[i] * 64;
    float s = 0.0f;
    #pragma unroll
    for (int c = 0; c < 64; ++c) s += a[c] * b[c];
    e[i] = s;
}

__global__ void s2s_ragg(const float* __restrict__ ev, const float* __restrict__ den,
                         const int* __restrict__ batch, const float* __restrict__ rg,
                         float* __restrict__ rbuf, int R) {
    int i = blockIdx.x * blockDim.x + threadIdx.x;
    if (i >= R * 64) return;
    int r = i >> 6, c = i & 63;
    int b = batch[r];
    float d = den[b];
    d = (d == 0.0f) ? 1.0f : d;
    atomicAdd(&rbuf[(size_t)b * 64 + c], (ev[r] / d) * rg[(size_t)r * 64 + c]);
}

__global__ void qstar_build(const float* __restrict__ hl, const float* __restrict__ rbuf,
                            float* __restrict__ q, int Bn) {
    int i = blockIdx.x * blockDim.x + threadIdx.x;
    if (i >= Bn * 128) return;
    int r = i >> 7, c = i & 127;
    q[i] = (c < 64) ? hl[(size_t)r * 64 + c] : rbuf[(size_t)r * 64 + (c - 64)];
}

__global__ void final_y(const float* __restrict__ fp, const float* __restrict__ W,
                        const float* __restrict__ b, float* __restrict__ y, int Bn) {
    int i = blockIdx.x * blockDim.x + threadIdx.x;
    if (i >= Bn) return;
    float s = b[0];
    #pragma unroll
    for (int c = 0; c < 64; ++c) s += fp[(size_t)i * 64 + c] * W[c];
    y[i] = s;
}

// ---------------------------------------------------------------------------
static inline dim3 gr1(long n, int bs = 256) { return dim3((unsigned)((n + bs - 1) / bs)); }

extern "C" void kernel_launch(void* const* d_in, const int* in_sizes, int n_in,
                              void* d_out, int out_size, void* d_ws, size_t ws_size,
                              hipStream_t stream) {
    const float* x          = (const float*)d_in[0];
    const float* edge_attr  = (const float*)d_in[1];
    const float* rg_x       = (const float*)d_in[2];
    const float* rg_eattr   = (const float*)d_in[3];
    const float* W_lin0     = (const float*)d_in[4];
    const float* b_lin0     = (const float*)d_in[5];
    const float* nn1_W1     = (const float*)d_in[6];
    const float* nn1_b1     = (const float*)d_in[7];
    const float* nn1_W2     = (const float*)d_in[8];
    const float* nn1_b2     = (const float*)d_in[9];
    const float* conv1_root = (const float*)d_in[10];
    const float* conv1_bias = (const float*)d_in[11];
    const float* gru1_Wih   = (const float*)d_in[12];
    const float* gru1_bih   = (const float*)d_in[13];
    const float* gru1_Whh   = (const float*)d_in[14];
    const float* gru1_bhh   = (const float*)d_in[15];
    const float* gat_W      = (const float*)d_in[16];
    const float* gat_asrc   = (const float*)d_in[17];
    const float* gat_adst   = (const float*)d_in[18];
    const float* gat_bias   = (const float*)d_in[19];
    const float* pgru_Wih   = (const float*)d_in[20];
    const float* pgru_bih   = (const float*)d_in[21];
    const float* pgru_Whh   = (const float*)d_in[22];
    const float* pgru_bhh   = (const float*)d_in[23];
    const float* nn2_W1     = (const float*)d_in[24];
    const float* nn2_b1     = (const float*)d_in[25];
    const float* nn2_W2     = (const float*)d_in[26];
    const float* nn2_b2     = (const float*)d_in[27];
    const float* conv2_root = (const float*)d_in[28];
    const float* conv2_bias = (const float*)d_in[29];
    const float* gru2_Wih   = (const float*)d_in[30];
    const float* gru2_bih   = (const float*)d_in[31];
    const float* gru2_Whh   = (const float*)d_in[32];
    const float* gru2_bhh   = (const float*)d_in[33];
    const float* lin_rg_W   = (const float*)d_in[34];
    const float* lin_rg_b   = (const float*)d_in[35];
    const float* lstm_Wih   = (const float*)d_in[36];
    const float* lstm_bih   = (const float*)d_in[37];
    const float* lstm_Whh   = (const float*)d_in[38];
    const float* lstm_bhh   = (const float*)d_in[39];
    const float* lin1_W     = (const float*)d_in[40];
    const float* lin1_b     = (const float*)d_in[41];
    const float* lin2_W     = (const float*)d_in[42];
    const float* lin2_b     = (const float*)d_in[43];
    const int*   ei         = (const int*)d_in[44];
    const int*   ptid       = (const int*)d_in[45];
    const int*   psid       = (const int*)d_in[46];
    const int*   rei        = (const int*)d_in[47];
    const int*   batch      = (const int*)d_in[48];

    const int N  = in_sizes[0] / 28;
    const int E  = in_sizes[1] / 4;
    const int R  = in_sizes[2] / 18;
    const int ER = in_sizes[3];
    const int* src1 = ei;       const int* dst1 = ei + E;
    const int* src2 = rei;      const int* dst2 = rei + ER;

    char* wb = (char*)d_ws;
    size_t off = 0;
    auto alloc = [&](size_t bytes) -> void* {
        void* p = wb + off;
        off += (bytes + 255) & ~(size_t)255;
        return p;
    };
    float*  f_h    = (float*)alloc((size_t)N * 64 * 4);
    float*  f_agg  = (float*)alloc((size_t)N * 64 * 4);
    float*  f_m    = (float*)alloc((size_t)N * 64 * 4);
    float*  f_cnt  = (float*)alloc((size_t)N * 4);
    float*  f_tmp  = (float*)alloc((size_t)N * 192 * 4);
    float*  f_tmp2 = (float*)alloc((size_t)N * 192 * 4);
    half_t* f_h1   = (half_t*)alloc((size_t)E * 128 * 2);
    half_t* f_h2   = (half_t*)alloc((size_t)ER * 64 * 2);
    float*  ws_ft  = (float*)alloc((size_t)N * 64 * 4);
    float*  wd_ft  = (float*)alloc((size_t)R * 64 * 4);
    float*  as_b   = (float*)alloc((size_t)N * 4);
    float*  ad_b   = (float*)alloc((size_t)R * 4);
    float*  plog   = (float*)alloc((size_t)N * 4);
    float*  pexp   = (float*)alloc((size_t)N * 4);
    float*  pmax   = (float*)alloc((size_t)R * 4);
    float*  pden   = (float*)alloc((size_t)R * 4);
    float*  rg0    = (float*)alloc((size_t)R * 64 * 4);
    float*  gbuf   = (float*)alloc((size_t)R * 64 * 4);
    float*  f_rg   = (float*)alloc((size_t)R * 64 * 4);
    float*  Apad   = (float*)alloc((size_t)R * 96 * 4);
    float*  qstar  = (float*)alloc((size_t)BGR * 128 * 4);
    float*  hlb    = (float*)alloc((size_t)BGR * 64 * 4);
    float*  clb    = (float*)alloc((size_t)BGR * 64 * 4);
    float*  bmax   = (float*)alloc((size_t)BGR * 4);
    float*  bden   = (float*)alloc((size_t)BGR * 4);
    float*  rbuf   = (float*)alloc((size_t)BGR * 64 * 4);
    float*  fpbuf  = (float*)alloc((size_t)BGR * 64 * 4);
    half_t* pT1    = (half_t*)alloc((size_t)64 * 129 * 64 * 2);
    half_t* pT2    = (half_t*)alloc((size_t)64 * 65 * 64 * 2);
    half_t* pRoot1 = (half_t*)alloc(64 * 64 * 2);
    half_t* pWih1  = (half_t*)alloc(64 * 192 * 2);
    half_t* pWhh1  = (half_t*)alloc(64 * 192 * 2);
    half_t* pGatW  = (half_t*)alloc(64 * 64 * 2);
    half_t* pPWih  = (half_t*)alloc(64 * 192 * 2);
    half_t* pPWhh  = (half_t*)alloc(64 * 192 * 2);
    half_t* pLinRG = (half_t*)alloc(96 * 64 * 2);
    half_t* pRoot2 = (half_t*)alloc(64 * 64 * 2);
    half_t* pWih2  = (half_t*)alloc(64 * 192 * 2);
    half_t* pWhh2  = (half_t*)alloc(64 * 192 * 2);
    half_t* pLWih  = (half_t*)alloc(128 * 256 * 2);
    half_t* pLWhh  = (half_t*)alloc(64 * 256 * 2);
    half_t* pLin1  = (half_t*)alloc(128 * 64 * 2);
    float*  y_out  = (float*)d_out;

    // ---- pack weights into fragment-major f16 ----
    pack_bfrag<<<gr1(64 * 64), 256, 0, stream>>>(conv1_root, 64, 64, 64, pRoot1);
    pack_bfrag<<<gr1(64 * 192), 256, 0, stream>>>(gru1_Wih, 64, 192, 64, pWih1);
    pack_bfrag<<<gr1(64 * 192), 256, 0, stream>>>(gru1_Whh, 64, 192, 64, pWhh1);
    pack_bfrag<<<gr1(64 * 64), 256, 0, stream>>>(gat_W, 64, 64, 64, pGatW);
    pack_bfrag<<<gr1(64 * 192), 256, 0, stream>>>(pgru_Wih, 64, 192, 64, pPWih);
    pack_bfrag<<<gr1(64 * 192), 256, 0, stream>>>(pgru_Whh, 64, 192, 64, pPWhh);
    pack_bfrag<<<gr1(96 * 64), 256, 0, stream>>>(lin_rg_W, 82, 64, 96, pLinRG);
    pack_bfrag<<<gr1(64 * 64), 256, 0, stream>>>(conv2_root, 64, 64, 64, pRoot2);
    pack_bfrag<<<gr1(64 * 192), 256, 0, stream>>>(gru2_Wih, 64, 192, 64, pWih2);
    pack_bfrag<<<gr1(64 * 192), 256, 0, stream>>>(gru2_Whh, 64, 192, 64, pWhh2);
    pack_bfrag<<<gr1(128 * 256), 256, 0, stream>>>(lstm_Wih, 128, 256, 128, pLWih);
    pack_bfrag<<<gr1(64 * 256), 256, 0, stream>>>(lstm_Whh, 64, 256, 64, pLWhh);
    pack_bfrag<<<gr1(128 * 64), 256, 0, stream>>>(lin1_W, 128, 64, 128, pLin1);
    pack_Tfrag<<<gr1((long)64 * 129 * 64), 256, 0, stream>>>(nn1_W2, nn1_b2, 128, pT1);
    pack_Tfrag<<<gr1((long)64 * 65 * 64), 256, 0, stream>>>(nn2_W2, nn2_b2, 64, pT2);

    // ---- lin0 ----
    lin0_relu<<<gr1((long)N * 64), 256, 0, stream>>>(x, W_lin0, b_lin0, f_h, N);

    // ---- edge MLP 1 + degrees ----
    edge_mlp1<<<gr1((long)E * 128), 256, 0, stream>>>(edge_attr, nn1_W1, nn1_b1, f_h1, E);
    fill_f32<<<gr1(N), 256, 0, stream>>>(f_cnt, 0.0f, N);
    deg_count<<<gr1(E), 256, 0, stream>>>(dst1, E, f_cnt);

    // ---- NNConv1 + GRU1, 2 passes ----
    for (int p = 0; p < 2; ++p) {
        fill_f32<<<gr1((long)N * 64), 256, 0, stream>>>(f_agg, 0.0f, N * 64);
        nnconv_msg<128><<<dim3((E + 63) / 64), 128, 0, stream>>>(f_h, f_h1, src1, dst1, E, pT1, f_agg);
        gemm_f16<<<dim3((N + 63) / 64), 128, 0, stream>>>(f_h, N, 64, 64, pRoot1, 64, conv1_bias, 0, f_tmp, 64);
        mean_root_relu<<<gr1((long)N * 64), 256, 0, stream>>>(f_agg, f_cnt, f_tmp, f_m, N * 64);
        gemm_f16<<<dim3((N + 63) / 64), 128, 0, stream>>>(f_m, N, 64, 64, pWih1, 192, gru1_bih, 0, f_tmp, 192);
        gemm_f16<<<dim3((N + 63) / 64), 128, 0, stream>>>(f_h, N, 64, 64, pWhh1, 192, gru1_bhh, 0, f_tmp2, 192);
        gru_elem<<<gr1((long)N * 64), 256, 0, stream>>>(f_tmp, f_tmp2, f_h, N, 0);
    }

    // ---- scatter-add pooling ----
    fill_f32<<<gr1((long)R * 64), 256, 0, stream>>>(rg0, 0.0f, R * 64);
    scatter_pool<<<gr1((long)N * 64), 256, 0, stream>>>(f_h, psid, ptid, N, rg0);

    // ---- GAT attention (loop-invariant) ----
    gemm_f16<<<dim3((N + 63) / 64), 128, 0, stream>>>(f_h, N, 64, 64, pGatW, 64, nullptr, 0, ws_ft, 64);
    gemm_f16<<<dim3((R + 63) / 64), 128, 0, stream>>>(rg0, R, 64, 64, pGatW, 64, nullptr, 0, wd_ft, 64);
    dot64<<<gr1(N), 256, 0, stream>>>(ws_ft, gat_asrc, as_b, N);
    dot64<<<gr1(R), 256, 0, stream>>>(wd_ft, gat_adst, ad_b, R);
    gat_logit<<<gr1(N), 256, 0, stream>>>(as_b, ad_b, psid, ptid, plog, N);
    fill_f32<<<gr1(R), 256, 0, stream>>>(pmax, -1e30f, R);
    seg_max_k<<<gr1(N), 256, 0, stream>>>(plog, ptid, pmax, N);
    fill_f32<<<gr1(R), 256, 0, stream>>>(pden, 0.0f, R);
    seg_exp_sum<<<gr1(N), 256, 0, stream>>>(plog, ptid, pmax, pden, pexp, N);
    fill_f32<<<gr1((long)R * 64), 256, 0, stream>>>(gbuf, 0.0f, R * 64);
    gat_agg<<<gr1((long)N * 64), 256, 0, stream>>>(pexp, pden, ptid, psid, ws_ft, gbuf, N);
    gat_bias_elu<<<gr1((long)R * 64), 256, 0, stream>>>(gbuf, gat_bias, R * 64);

    // ---- pool GRU x2 (attention output fixed -> gx fixed) ----
    copy_f32<<<gr1((long)R * 64), 256, 0, stream>>>(rg0, f_rg, R * 64);
    gemm_f16<<<dim3((R + 63) / 64), 128, 0, stream>>>(gbuf, R, 64, 64, pPWih, 192, pgru_bih, 0, f_tmp, 192);
    for (int p = 0; p < 2; ++p) {
        gemm_f16<<<dim3((R + 63) / 64), 128, 0, stream>>>(f_rg, R, 64, 64, pPWhh, 192, pgru_bhh, 0, f_tmp2, 192);
        gru_elem<<<gr1((long)R * 64), 256, 0, stream>>>(f_tmp, f_tmp2, f_rg, R, 1);
    }

    // ---- lin_rg on concat([rg, rg_x]) ----
    build_Apad<<<gr1((long)R * 96), 256, 0, stream>>>(f_rg, rg_x, Apad, R);
    gemm_f16<<<dim3((R + 63) / 64), 128, 0, stream>>>(Apad, R, 96, 96, pLinRG, 64, lin_rg_b, 0, f_rg, 64);

    // ---- NNConv2 + GRU2, 2 passes ----
    edge_mlp2<<<gr1((long)ER * 64), 256, 0, stream>>>(rg_eattr, nn2_W1, nn2_b1, f_h2, ER);
    fill_f32<<<gr1(R), 256, 0, stream>>>(f_cnt, 0.0f, R);
    deg_count<<<gr1(ER), 256, 0, stream>>>(dst2, ER, f_cnt);
    for (int p = 0; p < 2; ++p) {
        fill_f32<<<gr1((long)R * 64), 256, 0, stream>>>(f_agg, 0.0f, R * 64);
        nnconv_msg<64><<<dim3((ER + 63) / 64), 128, 0, stream>>>(f_rg, f_h2, src2, dst2, ER, pT2, f_agg);
        gemm_f16<<<dim3((R + 63) / 64), 128, 0, stream>>>(f_rg, R, 64, 64, pRoot2, 64, conv2_bias, 0, f_tmp, 64);
        mean_root_relu<<<gr1((long)R * 64), 256, 0, stream>>>(f_agg, f_cnt, f_tmp, f_m, R * 64);
        gemm_f16<<<dim3((R + 63) / 64), 128, 0, stream>>>(f_m, R, 64, 64, pWih2, 192, gru2_bih, 0, f_tmp, 192);
        gemm_f16<<<dim3((R + 63) / 64), 128, 0, stream>>>(f_rg, R, 64, 64, pWhh2, 192, gru2_bhh, 0, f_tmp2, 192);
        gru_elem<<<gr1((long)R * 64), 256, 0, stream>>>(f_tmp, f_tmp2, f_rg, R, 0);
    }

    // ---- Set2Set readout, 3 steps ----
    fill_f32<<<gr1(BGR * 128), 256, 0, stream>>>(qstar, 0.0f, BGR * 128);
    fill_f32<<<gr1(BGR * 64), 256, 0, stream>>>(hlb, 0.0f, BGR * 64);
    fill_f32<<<gr1(BGR * 64), 256, 0, stream>>>(clb, 0.0f, BGR * 64);
    for (int it = 0; it < 3; ++it) {
        gemm_f16<<<dim3(2), 128, 0, stream>>>(qstar, BGR, 128, 128, pLWih, 256, lstm_bih, 0, f_tmp, 256);
        gemm_f16<<<dim3(2), 128, 0, stream>>>(hlb, BGR, 64, 64, pLWhh, 256, lstm_bhh, 0, f_tmp2, 256);
        lstm_elem<<<gr1(BGR * 64), 256, 0, stream>>>(f_tmp, f_tmp2, hlb, clb, BGR);
        s2s_edot<<<gr1(R), 256, 0, stream>>>(f_rg, hlb, batch, plog, R);
        fill_f32<<<gr1(BGR), 256, 0, stream>>>(bmax, -1e30f, BGR);
        seg_max_k<<<gr1(R), 256, 0, stream>>>(plog, batch, bmax, R);
        fill_f32<<<gr1(BGR), 256, 0, stream>>>(bden, 0.0f, BGR);
        seg_exp_sum<<<gr1(R), 256, 0, stream>>>(plog, batch, bmax, bden, pexp, R);
        fill_f32<<<gr1(BGR * 64), 256, 0, stream>>>(rbuf, 0.0f, BGR * 64);
        s2s_ragg<<<gr1((long)R * 64), 256, 0, stream>>>(pexp, bden, batch, f_rg, rbuf, R);
        qstar_build<<<gr1(BGR * 128), 256, 0, stream>>>(hlb, rbuf, qstar, BGR);
    }

    // ---- heads ----
    gemm_f16<<<dim3(2), 128, 0, stream>>>(qstar, BGR, 128, 128, pLin1, 64, lin1_b, 1, fpbuf, 64);
    final_y<<<gr1(BGR), 256, 0, stream>>>(fpbuf, lin2_W, lin2_b, y_out, BGR);
    copy_f32<<<gr1(BGR * 64), 256, 0, stream>>>(fpbuf, y_out + BGR, BGR * 64);

    (void)n_in; (void)out_size; (void)ws_size;
}
